// NoMlpGpt2Attention_72859825210048
// MI455X (gfx1250) — matrix operand order repaired and checked by reading, not compiled, not run on hardware
//
#include <hip/hip_runtime.h>
#include <hip/hip_bf16.h>
#include <stdint.h>

// Problem constants (match reference).
#define B_  2
#define S_  2048
#define E_  1024
#define H_  16
#define D_  64
#define BS_ (B_*S_)   // 4096

typedef __attribute__((ext_vector_type(16))) __bf16 v16bf;
typedef __attribute__((ext_vector_type(8)))  float  v8f;

// ---- CDNA5 feature probes (compile-safe: guarded fallbacks everywhere) ----
#if __has_builtin(__builtin_amdgcn_global_load_async_to_lds_b128)
#define HAVE_ASYNC 1
#else
#define HAVE_ASYNC 0
#endif
#if __has_builtin(__builtin_amdgcn_tensor_load_to_lds)
#define HAVE_TDM 1
#else
#define HAVE_TDM 0
#endif

__device__ __forceinline__ __bf16 f2bf(float f) {
  unsigned u = __builtin_bit_cast(unsigned, f);
  u += 0x7FFFu + ((u >> 16) & 1u);          // round-to-nearest-even
  unsigned short h = (unsigned short)(u >> 16);
  return __builtin_bit_cast(__bf16, h);
}

__device__ __forceinline__ v8f zero8() {
  v8f z = {0.f,0.f,0.f,0.f,0.f,0.f,0.f,0.f};
  return z;
}

// ---------------------------------------------------------------------------
// Staging primitives: async global->LDS (ASYNCcnt) with sync fallback.
// 16 bytes per lane, 16B-aligned on both sides.
// Builtin signature (from probe diagnostics): 4 args,
//   (v4i32 addrspace(1)*, v4i32 addrspace(3)*, offset, cpol)
// ---------------------------------------------------------------------------
__device__ __forceinline__ void stage16(__bf16* lds, const __bf16* g) {
#if defined(__HIP_DEVICE_COMPILE__) && HAVE_ASYNC
  typedef int v4i __attribute__((vector_size(16)));
  typedef __attribute__((address_space(1))) v4i as1_v4i;
  typedef __attribute__((address_space(3))) v4i as3_v4i;
  as1_v4i* gp = __builtin_bit_cast(as1_v4i*, (size_t)g);
  as3_v4i* lp = __builtin_bit_cast(as3_v4i*, (unsigned)(size_t)lds);
  __builtin_amdgcn_global_load_async_to_lds_b128(gp, lp, 0, 0);
#else
  *(uint4*)lds = *(const uint4*)g;
#endif
}

__device__ __forceinline__ void stage_fence() {
#if defined(__HIP_DEVICE_COMPILE__) && HAVE_ASYNC
#if __has_builtin(__builtin_amdgcn_s_wait_asynccnt)
  __builtin_amdgcn_s_wait_asynccnt(0);
#else
  asm volatile("s_wait_asynccnt 0x0" ::: "memory");
#endif
#endif
}

__device__ __forceinline__ void wait_tensor0() {
#if defined(__HIP_DEVICE_COMPILE__) && __has_builtin(__builtin_amdgcn_s_wait_tensorcnt)
  __builtin_amdgcn_s_wait_tensorcnt(0);
#endif
}

// ---------------------------------------------------------------------------
// TDM: 2D tile DMA global->LDS. D# per CDNA5 ISA ch.8: group0 = {ctl, lds,
// gaddr, type=2}, group1 = {data_size=2B, tensor dims, tile dims, stride}.
// 6-arg builtin form: (u32x4, i32x8, i32x4, i32x4, i32x8, i32 cpol).
// ---------------------------------------------------------------------------
#if HAVE_TDM
typedef unsigned __attribute__((ext_vector_type(4))) u32x4;
typedef int      __attribute__((ext_vector_type(8))) i32x8;
typedef int      __attribute__((ext_vector_type(4))) i32x4;
#endif

__device__ __forceinline__ void tdm_load_2d(__bf16* lds, const __bf16* gtile,
                                            unsigned tensor_w, unsigned tensor_h,
                                            unsigned tile_w, unsigned tile_h,
                                            unsigned row_stride_elems) {
#if defined(__HIP_DEVICE_COMPILE__) && HAVE_TDM
  unsigned long long ga = (unsigned long long)(size_t)gtile;
  u32x4 g0;
  g0.x = 1u;                                   // count=1, user descriptor
  g0.y = (unsigned)(size_t)lds;                // LDS byte address (addr[31:0])
  g0.z = (unsigned)ga;                         // global_addr[95:64]
  g0.w = (unsigned)(ga >> 32) | (2u << 30);    // global_addr[120:96], type=2
  i32x8 g1;
  g1[0] = (int)(1u << 16);                     // data_size=1 -> 2 bytes
  g1[1] = (int)(tensor_w << 16);               // tensor_dim0[15:0]
  g1[2] = (int)((tensor_w >> 16) | (tensor_h << 16));  // td0 hi | td1 lo
  g1[3] = (int)((tensor_h >> 16) | (tile_w << 16));    // td1 hi | tile_dim0
  g1[4] = (int)(tile_h & 0xFFFFu);             // tile_dim1 (tile_dim2=0)
  g1[5] = (int)row_stride_elems;               // tensor_dim0_stride lo
  g1[6] = 0; g1[7] = 0;
  i32x4 z4 = {0, 0, 0, 0};
  i32x8 z8 = {0, 0, 0, 0, 0, 0, 0, 0};
  __builtin_amdgcn_tensor_load_to_lds(g0, g1, z4, z4, z8, 0);
#else
  (void)lds; (void)gtile; (void)tensor_w; (void)tensor_h;
  (void)tile_w; (void)tile_h; (void)row_stride_elems;
#endif
}

// ---------------------------------------------------------------------------
// WMMA fragment helpers (layouts per ISA 7.12.2).
// ---------------------------------------------------------------------------
__device__ __forceinline__ v16bf load_afrag(const __bf16* row, int half) {
  union { v16bf v; unsigned u[8]; } r;
  const unsigned* p = (const unsigned*)row;
#pragma unroll
  for (int v = 0; v < 8; ++v) {
    int k0 = (v < 4 ? 2*v : 16 + 2*(v-4)) + 8*half;   // even
    r.u[v] = p[k0 >> 1];
  }
  return r.v;
}

__device__ __forceinline__ v16bf load_bfrag(const __bf16* row, int half) {
  union { v16bf v; unsigned u[8]; } r;
  const unsigned* p = (const unsigned*)row;
#pragma unroll
  for (int v = 0; v < 8; ++v) r.u[v] = p[((v < 4 ? 2*v : 16 + 2*(v-4)) + 8*half) >> 1];
  return r.v;
}

__device__ __forceinline__ v8f wmma_bf16(v16bf a, v16bf b, v8f c) {
  return __builtin_amdgcn_wmma_f32_16x16x32_bf16(false, a, false, b,
                                                 (short)0, c, false, false);
}

// ---------------------------------------------------------------------------
// Stage 0: conversions to bf16 (and weight transposition to [N][K]).
// ---------------------------------------------------------------------------
__global__ void k_cvt(const float* __restrict__ src, __bf16* __restrict__ dst,
                      int n) {
  int i = blockIdx.x * blockDim.x + threadIdx.x;
  if (i < n) dst[i] = f2bf(src[i]);
}

__global__ void k_transpose_cvt(const float* __restrict__ src,
                                __bf16* __restrict__ dst, int K, int N) {
  int i = blockIdx.x * blockDim.x + threadIdx.x;
  if (i < K * N) {
    int n = i / K, k = i - n * K;
    dst[i] = f2bf(src[k * N + n]);
  }
}

// ---------------------------------------------------------------------------
// Stage 1: QKV projection. C[4096,3072] = X[4096,1024] x W. A/B tiles staged
// by TDM when available (wave 0 issues two descriptors, waits TENSORcnt).
// ---------------------------------------------------------------------------
__global__ __launch_bounds__(256) void k_qkv_gemm(
    const __bf16* __restrict__ X, const __bf16* __restrict__ Wt,
    const float* __restrict__ bias, const int* __restrict__ layer_idx,
    __bf16* __restrict__ qb, __bf16* __restrict__ kb, __bf16* __restrict__ vb) {
  __shared__ __align__(16) __bf16 As[128 * 32];
  __shared__ __align__(16) __bf16 Bs[64 * 32];
  const int tid = threadIdx.x, lane = tid & 31, wave = tid >> 5;
  const int half = lane >> 4, l15 = lane & 15;
  const int mblock = blockIdx.y * 128, nblock = blockIdx.x * 64;
  const float scale = rsqrtf((float)D_) / (float)(*layer_idx + 1);

  v8f acc[4];
#pragma unroll
  for (int i = 0; i < 4; ++i) acc[i] = zero8();

  for (int kt = 0; kt < E_; kt += 32) {
#if HAVE_TDM
    if (wave == 0) {   // one wave drives the tile DMAs (EXEC-independent)
      tdm_load_2d(As, X + (size_t)mblock * E_ + kt, E_, BS_, 32, 128, E_);
      tdm_load_2d(Bs, Wt + (size_t)nblock * E_ + kt, E_, 3 * E_, 32, 64, E_);
      wait_tensor0();
    }
#else
    {  // stage A: 128 rows x 32 K
      int row = tid >> 1, part = tid & 1;
      const __bf16* s = X + (size_t)(mblock + row) * E_ + kt + part * 16;
      __bf16* d = As + row * 32 + part * 16;
      stage16(d, s); stage16(d + 8, s + 8);
      if (kt + 32 < E_) __builtin_prefetch(s + 32, 0, 0);
    }
    {  // stage B: 64 N-rows x 32 K
      int row = tid >> 2, part = tid & 3;
      stage16(Bs + row * 32 + part * 8,
              Wt + (size_t)(nblock + row) * E_ + kt + part * 8);
    }
    stage_fence();
#endif
    __syncthreads();
    v16bf a = load_afrag(&As[(wave * 16 + l15) * 32], half);
#pragma unroll
    for (int nt = 0; nt < 4; ++nt) {
      v16bf b = load_bfrag(&Bs[(nt * 16 + l15) * 32], half);
      acc[nt] = wmma_bf16(a, b, acc[nt]);
    }
    __syncthreads();
  }

  const int mrow = mblock + wave * 16;
#pragma unroll
  for (int nt = 0; nt < 4; ++nt) {
    int n = nblock + nt * 16 + l15;
    int sel = n >> 10, rem = n & 1023;   // {q,k,v}, h*64+d
    float bn = bias[n];
#pragma unroll
    for (int r = 0; r < 8; ++r) {
      int m = mrow + r + 8 * half;
      int b = m >> 11, s = m & (S_ - 1);
      float v = acc[nt][r] + bn;
      if (sel == 0) {
        qb[(((size_t)b * H_ + (rem >> 6)) * S_ + s) * D_ + (rem & 63)] = f2bf(v * scale);
      } else if (sel == 1) {
        kb[(((size_t)b * H_ + (rem >> 6)) * S_ + s) * D_ + (rem & 63)] = f2bf(v);
      } else {
        vb[(size_t)m * E_ + rem] = f2bf(v);
      }
    }
  }
}

// ---------------------------------------------------------------------------
// Stage 2: fused gate/ff GEMMs: vg = relu(V x Wg) * (V x Wf) -> [B,H,S,D].
// ---------------------------------------------------------------------------
__global__ __launch_bounds__(256) void k_gateff_gemm(
    const __bf16* __restrict__ V, const __bf16* __restrict__ Wgt,
    const __bf16* __restrict__ Wft, __bf16* __restrict__ vg) {
  __shared__ __align__(16) __bf16 As[128 * 32];
  __shared__ __align__(16) __bf16 Bg[64 * 32];
  __shared__ __align__(16) __bf16 Bf[64 * 32];
  const int tid = threadIdx.x, lane = tid & 31, wave = tid >> 5;
  const int half = lane >> 4, l15 = lane & 15;
  const int mblock = blockIdx.y * 128, nblock = blockIdx.x * 64;

  v8f ag[4], af[4];
#pragma unroll
  for (int i = 0; i < 4; ++i) { ag[i] = zero8(); af[i] = zero8(); }

  for (int kt = 0; kt < E_; kt += 32) {
    {
      int row = tid >> 1, part = tid & 1;
      const __bf16* s = V + (size_t)(mblock + row) * E_ + kt + part * 16;
      __bf16* d = As + row * 32 + part * 16;
      stage16(d, s); stage16(d + 8, s + 8);
      if (kt + 32 < E_) __builtin_prefetch(s + 32, 0, 0);
    }
    {
      int row = tid >> 2, part = tid & 3;
      size_t so = (size_t)(nblock + row) * E_ + kt + part * 8;
      stage16(Bg + row * 32 + part * 8, Wgt + so);
      stage16(Bf + row * 32 + part * 8, Wft + so);
    }
    stage_fence();
    __syncthreads();
    v16bf a = load_afrag(&As[(wave * 16 + l15) * 32], half);
#pragma unroll
    for (int nt = 0; nt < 4; ++nt) {
      ag[nt] = wmma_bf16(a, load_bfrag(&Bg[(nt * 16 + l15) * 32], half), ag[nt]);
      af[nt] = wmma_bf16(a, load_bfrag(&Bf[(nt * 16 + l15) * 32], half), af[nt]);
    }
    __syncthreads();
  }

  const int mrow = mblock + wave * 16;
#pragma unroll
  for (int nt = 0; nt < 4; ++nt) {
    int n = nblock + nt * 16 + l15;
    int h = n >> 6, d = n & 63;
#pragma unroll
    for (int r = 0; r < 8; ++r) {
      int m = mrow + r + 8 * half;
      int b = m >> 11, s = m & (S_ - 1);
      float g = ag[nt][r];
      float v = (g > 0.f ? g : 0.f) * af[nt][r];
      vg[(((size_t)b * H_ + h) * S_ + s) * D_ + d] = f2bf(v);
    }
  }
}

// ---------------------------------------------------------------------------
// Stage 3: causal flash attention (one block per (b,h,128 q-rows)).
// ---------------------------------------------------------------------------
__global__ __launch_bounds__(256) void k_flash(
    const __bf16* __restrict__ qb, const __bf16* __restrict__ kb,
    const __bf16* __restrict__ vg, __bf16* __restrict__ ao /*[BS,E]*/) {
  __shared__ __align__(16) __bf16 Ks[32 * 64];       // [t][d]
  __shared__ __align__(16) __bf16 Vt[64 * 32];       // [d][t]
  __shared__ __align__(16) __bf16 Ps[8 * 16 * 32];   // per-wave P patch
  const int tid = threadIdx.x, lane = tid & 31, wave = tid >> 5;
  const int half = lane >> 4, l15 = lane & 15;
  const int b = blockIdx.z, h = blockIdx.y, qbase = blockIdx.x * 128;
  const size_t bh = (size_t)b * H_ + h;
  const int qrow0 = qbase + wave * 16;

  v16bf aq[2];  // q A-fragments, D split 2x32
#pragma unroll
  for (int kd = 0; kd < 2; ++kd)
    aq[kd] = load_afrag(qb + (bh * S_ + qrow0 + l15) * D_ + kd * 32, half);

  v8f acc[4];
#pragma unroll
  for (int i = 0; i < 4; ++i) acc[i] = zero8();
  float rm[8], rl[8];
#pragma unroll
  for (int r = 0; r < 8; ++r) { rm[r] = -3.0e38f; rl[r] = 0.f; }

  for (int tbase = 0; tbase < qbase + 128; tbase += 32) {
    {  // stage K: 32 rows x 64 d (async when available)
      int row = tid >> 3, part = tid & 7;
      stage16(Ks + row * 64 + part * 8,
              kb + (bh * S_ + tbase + row) * D_ + part * 8);
    }
    {  // stage V transposed: coalesced read, transposed LDS write
      int tt = tid >> 3, dpart = (tid & 7) * 8;
      union { uint4 q; __bf16 e[8]; } u;
      u.q = *(const uint4*)(vg + (bh * S_ + tbase + tt) * D_ + dpart);
#pragma unroll
      for (int j = 0; j < 8; ++j) Vt[(dpart + j) * 32 + tt] = u.e[j];
    }
    stage_fence();
    __syncthreads();

    if (tbase <= qrow0 + 15) {  // wave-uniform causal skip
      v8f sc[2];
#pragma unroll
      for (int st = 0; st < 2; ++st) {
        v8f s = zero8();
#pragma unroll
        for (int kd = 0; kd < 2; ++kd)
          s = wmma_bf16(aq[kd],
                        load_bfrag(&Ks[(st * 16 + l15) * 64 + kd * 32], half), s);
        int tcol = tbase + st * 16 + l15;
#pragma unroll
        for (int r = 0; r < 8; ++r) {
          int qr = qrow0 + r + 8 * half;
          if (tcol > qr) s[r] = -1.0e9f;   // causal mask
        }
        sc[st] = s;
      }
      __bf16* Pw = Ps + wave * 512;
#pragma unroll
      for (int r = 0; r < 8; ++r) {
        float mx = fmaxf(sc[0][r], sc[1][r]);
#pragma unroll
        for (int off = 8; off >= 1; off >>= 1)
          mx = fmaxf(mx, __shfl_xor(mx, off, 32));
        float mnew = fmaxf(rm[r], mx);
        float alpha = __expf(rm[r] - mnew);
        float p0 = __expf(sc[0][r] - mnew);
        float p1 = __expf(sc[1][r] - mnew);
        float ps = p0 + p1;
#pragma unroll
        for (int off = 8; off >= 1; off >>= 1) ps += __shfl_xor(ps, off, 32);
        rl[r] = rl[r] * alpha + ps;
        rm[r] = mnew;
#pragma unroll
        for (int nt = 0; nt < 4; ++nt) acc[nt][r] *= alpha;
        __bf16* prow = Pw + (r + 8 * half) * 32;
        prow[l15] = f2bf(p0);
        prow[16 + l15] = f2bf(p1);
      }
      v16bf ap = load_afrag(Pw + l15 * 32, half);
#pragma unroll
      for (int nt = 0; nt < 4; ++nt)
        acc[nt] = wmma_bf16(ap, load_bfrag(&Vt[(nt * 16 + l15) * 32], half),
                            acc[nt]);
    }
    __syncthreads();
  }

#pragma unroll
  for (int r = 0; r < 8; ++r) {
    float inv = 1.0f / rl[r];
    int qr = qrow0 + r + 8 * half;
#pragma unroll
    for (int nt = 0; nt < 4; ++nt)
      ao[((size_t)b * S_ + qr) * E_ + h * D_ + nt * 16 + l15] =
          f2bf(acc[nt][r] * inv);
  }
}

// ---------------------------------------------------------------------------
// Stage 4: output projection, f32 result + bias.
// ---------------------------------------------------------------------------
__global__ __launch_bounds__(256) void k_proj_gemm(
    const __bf16* __restrict__ A, const __bf16* __restrict__ Wt,
    const float* __restrict__ bias, float* __restrict__ out) {
  __shared__ __align__(16) __bf16 As[128 * 32];
  __shared__ __align__(16) __bf16 Bs[64 * 32];
  const int tid = threadIdx.x, lane = tid & 31, wave = tid >> 5;
  const int half = lane >> 4, l15 = lane & 15;
  const int mblock = blockIdx.y * 128, nblock = blockIdx.x * 64;

  v8f acc[4];
#pragma unroll
  for (int i = 0; i < 4; ++i) acc[i] = zero8();

  for (int kt = 0; kt < E_; kt += 32) {
    {
      int row = tid >> 1, part = tid & 1;
      const __bf16* s = A + (size_t)(mblock + row) * E_ + kt + part * 16;
      __bf16* d = As + row * 32 + part * 16;
      stage16(d, s); stage16(d + 8, s + 8);
      if (kt + 32 < E_) __builtin_prefetch(s + 32, 0, 0);
    }
    {
      int row = tid >> 2, part = tid & 3;
      stage16(Bs + row * 32 + part * 8,
              Wt + (size_t)(nblock + row) * E_ + kt + part * 8);
    }
    stage_fence();
    __syncthreads();
    v16bf a = load_afrag(&As[(wave * 16 + l15) * 32], half);
#pragma unroll
    for (int nt = 0; nt < 4; ++nt)
      acc[nt] = wmma_bf16(a, load_bfrag(&Bs[(nt * 16 + l15) * 32], half),
                          acc[nt]);
    __syncthreads();
  }

  const int mrow = mblock + wave * 16;
#pragma unroll
  for (int nt = 0; nt < 4; ++nt) {
    int n = nblock + nt * 16 + l15;
    float bn = bias[n];
#pragma unroll
    for (int r = 0; r < 8; ++r) {
      int m = mrow + r + 8 * half;
      out[(size_t)m * E_ + n] = acc[nt][r] + bn;
    }
  }
}

// ---------------------------------------------------------------------------
extern "C" void kernel_launch(void* const* d_in, const int* in_sizes, int n_in,
                              void* d_out, int out_size, void* d_ws,
                              size_t ws_size, hipStream_t stream) {
  const float* hidden   = (const float*)d_in[0];
  // d_in[1]: dense causal mask — applied analytically in k_flash.
  const int*   layer_ix = (const int*)d_in[2];
  const float* W_attn   = (const float*)d_in[3];
  const float* b_attn   = (const float*)d_in[4];
  const float* W_proj   = (const float*)d_in[5];
  const float* b_proj   = (const float*)d_in[6];
  const float* W_v_ff   = (const float*)d_in[7];
  const float* W_v_gate = (const float*)d_in[8];
  (void)in_sizes; (void)n_in; (void)out_size; (void)ws_size;

  char* ws = (char*)d_ws;
  size_t ofs = 0;
  auto carve = [&](size_t bytes) {
    char* p = ws + ofs;
    ofs += (bytes + 255) & ~(size_t)255;
    return p;
  };
  __bf16* Xbf  = (__bf16*)carve((size_t)BS_ * E_ * 2);
  __bf16* Wqkv = (__bf16*)carve((size_t)3 * E_ * E_ * 2);
  __bf16* Wg   = (__bf16*)carve((size_t)E_ * E_ * 2);
  __bf16* Wf   = (__bf16*)carve((size_t)E_ * E_ * 2);
  __bf16* Wp   = (__bf16*)carve((size_t)E_ * E_ * 2);
  __bf16* qb   = (__bf16*)carve((size_t)BS_ * E_ * 2);
  __bf16* kb   = (__bf16*)carve((size_t)BS_ * E_ * 2);
  __bf16* vb   = (__bf16*)carve((size_t)BS_ * E_ * 2);
  __bf16* vg   = (__bf16*)carve((size_t)BS_ * E_ * 2);
  __bf16* ao   = (__bf16*)carve((size_t)BS_ * E_ * 2);

  dim3 blk(256);
  k_cvt<<<(BS_ * E_) / 256, blk, 0, stream>>>(hidden, Xbf, BS_ * E_);
  k_transpose_cvt<<<(3 * E_ * E_) / 256, blk, 0, stream>>>(W_attn, Wqkv, E_, 3 * E_);
  k_transpose_cvt<<<(E_ * E_) / 256, blk, 0, stream>>>(W_v_gate, Wg, E_, E_);
  k_transpose_cvt<<<(E_ * E_) / 256, blk, 0, stream>>>(W_v_ff, Wf, E_, E_);
  k_transpose_cvt<<<(E_ * E_) / 256, blk, 0, stream>>>(W_proj, Wp, E_, E_);

  k_qkv_gemm<<<dim3(48, 32), blk, 0, stream>>>(Xbf, Wqkv, b_attn, layer_ix,
                                               qb, kb, vb);
  k_gateff_gemm<<<dim3(16, 32), blk, 0, stream>>>(vb, Wg, Wf, vg);
  k_flash<<<dim3(S_ / 128, H_, B_), blk, 0, stream>>>(qb, kb, vg, ao);
  k_proj_gemm<<<dim3(16, 32), blk, 0, stream>>>(ao, Wp, b_proj, (float*)d_out);
}
